// MRGNN_31104153157793
// MI455X (gfx1250) — compile-verified, matrix operand order checked by reading, not run on hardware
//
#include <hip/hip_runtime.h>
#include <hip/hip_bf16.h>

typedef __attribute__((ext_vector_type(16))) __bf16 v16bf;
typedef __attribute__((ext_vector_type(8)))  float  v8f;

#define NN   8192
#define EE   262144
#define INC  64
#define KHOP 3
#define DIN  192          // IN_C*K
#define DOUT 192          // OUT_C*K
#define GG   64
#define NCL  10
#define BNEPS 1e-5f
#define NEG_BIG -3.402823466e38f

// ---------------- small utility kernels ----------------
__global__ void k_zero(float* p, int n) {
    int i = blockIdx.x * blockDim.x + threadIdx.x;
    if (i < n) p[i] = 0.f;
}
__global__ void k_copy(const float* __restrict__ s, float* __restrict__ d, int n) {
    int i = blockIdx.x * blockDim.x + threadIdx.x;
    if (i < n) d[i] = s[i];
}

// ---------------- graph: degree, dinv, SpMM ----------------
__global__ void k_degree(const long long* __restrict__ ei, float* __restrict__ deg) {
    int e = blockIdx.x * blockDim.x + threadIdx.x;
    if (e < EE) atomicAdd(&deg[(int)ei[e]], 1.0f);
}
__global__ void k_dinv(float* deg) {   // in-place deg -> dinv
    int i = blockIdx.x * blockDim.x + threadIdx.x;
    if (i < NN) {
        float d = deg[i];
        deg[i] = (d > 0.f) ? rsqrtf(fmaxf(d, 1.0f)) : 0.f;
    }
}
// dst (pre-initialized to src, covering the +I diagonal) += vals * src[col]
__global__ void k_spmm(const long long* __restrict__ ei, const float* __restrict__ dinv,
                       const float* __restrict__ src, float* __restrict__ dst) {
    int e = blockIdx.x * blockDim.x + threadIdx.x;
    if (e >= EE) return;
    int r = (int)ei[e];
    int c = (int)ei[EE + e];
    float v = -dinv[r] * dinv[c];
    const float* s = src + (size_t)c * INC;
    float*       d = dst + (size_t)r * INC;
#pragma unroll 8
    for (int ch = 0; ch < INC; ++ch) atomicAdd(&d[ch], v * s[ch]);
}

// ---------------- spectral norm (one power-iteration step) ----------------
__global__ void k_specnorm(const float* __restrict__ W, const float* __restrict__ u,
                           float* __restrict__ invsig) {
    __shared__ float red[DIN];
    __shared__ float vv[DIN];
    int t = threadIdx.x;              // 0..191
    float acc = 0.f;
    for (int i = 0; i < DOUT; ++i) acc += W[i * DIN + t] * u[i];
    red[t] = acc * acc;
    __syncthreads();
    if (t < 64) red[t] += red[t + 64] + red[t + 128];
    __syncthreads();
    for (int s = 32; s > 0; s >>= 1) { if (t < s) red[t] += red[t + s]; __syncthreads(); }
    float nrm = sqrtf(red[0]) + 1e-12f;
    vv[t] = acc / nrm;
    __syncthreads();
    float acc2 = 0.f;
    for (int k = 0; k < DIN; ++k) acc2 += W[t * DIN + k] * vv[k];
    red[t] = acc2 * u[t];
    __syncthreads();
    if (t < 64) red[t] += red[t + 64] + red[t + 128];
    __syncthreads();
    for (int s = 32; s > 0; s >>= 1) { if (t < s) red[t] += red[t + s]; __syncthreads(); }
    if (t == 0) invsig[0] = 1.0f / red[0];
}

// masked + spectral-scaled weight -> bf16 (row-major [DOUT][DIN])
__global__ void k_build_wbf(const float* __restrict__ W, const float* __restrict__ invsig,
                            __bf16* __restrict__ Wbf) {
    int idx = blockIdx.x * blockDim.x + threadIdx.x;
    if (idx >= DOUT * DIN) return;
    int n = idx / DIN, k = idx % DIN;
    int blk = n / 64;                       // OUT_C = 64
    float w = (k < 64 * (blk + 1)) ? W[idx] * invsig[0] : 0.f;
    Wbf[idx] = (__bf16)w;
}

// Hcat = [X | LX | L^2X] -> bf16 row-major [NN][DIN]
__global__ void k_cat_bf16(const float* __restrict__ x, const float* __restrict__ h1,
                           const float* __restrict__ h2, __bf16* __restrict__ A) {
    int idx = blockIdx.x * blockDim.x + threadIdx.x;
    if (idx >= NN * DIN) return;
    int n = idx / DIN, c = idx % DIN;
    float v;
    if      (c < 64)  v = x [(size_t)n * 64 + c];
    else if (c < 128) v = h1[(size_t)n * 64 + (c - 64)];
    else              v = h2[(size_t)n * 64 + (c - 128)];
    A[idx] = (__bf16)v;
}

// ---------------- WMMA GEMM: H[m,n] = sum_k A[m,k]*Wbf[n,k] + bias[n] ----------------
// Block = 4 waves = one tn column-panel x 4 tm tiles. The 16x192 bf16 B panel
// (6 KB) is staged once per block into LDS with GLOBAL_LOAD_ASYNC_TO_LDS_B128
// (ASYNCcnt path), then all 4 waves read their fragments from LDS.
#define TILES_N 12          // 192/16
#define WPB     4           // waves per block
__global__ void k_gemm_wmma(const __bf16* __restrict__ A, const __bf16* __restrict__ B,
                            const float* __restrict__ bias, float* __restrict__ H) {
    __shared__ __attribute__((aligned(16))) __bf16 Bs[16 * DIN];   // 6144 bytes

    int lane = threadIdx.x & 31;
    int wave = threadIdx.x >> 5;
    int tn = blockIdx.x % TILES_N;
    int tm = (blockIdx.x / TILES_N) * WPB + wave;

    // ---- async-stage B panel: 16 rows x 192 bf16 = 384 x 16-byte chunks ----
    {
        unsigned ldsbase = (unsigned)(size_t)&Bs[0];     // LDS byte address (low 32 bits)
        const __bf16* gsrc = B + (size_t)(tn * 16) * DIN; // contiguous 6144B panel
#pragma unroll
        for (int it = 0; it < 3; ++it) {                 // 3 * 128 threads = 384
            int t = it * 128 + threadIdx.x;
            unsigned dst = ldsbase + t * 16;
            const __bf16* src = gsrc + t * 8;            // 8 bf16 = 16 bytes
            asm volatile("global_load_async_to_lds_b128 %0, %1, off"
                         :: "v"(dst), "v"(src) : "memory");
        }
        asm volatile("s_wait_asynccnt 0x0" ::: "memory");
    }
    __syncthreads();

    int r = lane & 15;      // A-row within tile / B-col within panel / D column
    int h = lane >> 4;      // lane-half selector
    const __bf16* Arow = A + (size_t)(tm * 16 + r) * DIN;
    const __bf16* Brow = &Bs[r * DIN];                   // LDS-resident B column n
    v8f acc = {};
#pragma unroll
    for (int kb = 0; kb < DIN; kb += 32) {
        __builtin_prefetch(Arow + kb + 32, 0, 0);        // global_prefetch_b8
        v16bf a, b;
#pragma unroll
        for (int e = 0; e < 16; ++e) {
            int i = e >> 1, pos = e & 1;
            // ISA 16-bit A layout: vgpr 0-3 -> K=0..7(+8h), vgpr 4-7 -> K=16..23(+8h)
            int kk = ((i < 4) ? (2 * i) : (16 + 2 * (i - 4))) + 8 * h + pos;
            a[e] = Arow[kb + kk];
            b[e] = Brow[kb + kk];                        // ds_load path
        }
        acc = __builtin_amdgcn_wmma_f32_16x16x32_bf16(
                  false, a, false, b, (short)0, acc, false, false);
    }
    int n = tn * 16 + r;
    float bi = bias[n];
#pragma unroll
    for (int j = 0; j < 8; ++j) {
        int m = tm * 16 + j + 8 * h;    // ISA C/D layout: vgpr j -> M=j (+8 for hi half)
        H[(size_t)m * DOUT + n] = acc[j] + bi;
    }
}

// ---------------- batchnorm stats over rows (per channel) ----------------
__global__ void k_bn_stats(const float* __restrict__ H, float* __restrict__ mu,
                           float* __restrict__ var, int rows, int cols) {
    int c = blockIdx.x;
    __shared__ float s1[256], s2[256];
    float a = 0.f, b = 0.f;
    for (int rr = threadIdx.x; rr < rows; rr += 256) {
        float v = H[(size_t)rr * cols + c];
        a += v; b += v * v;
    }
    s1[threadIdx.x] = a; s2[threadIdx.x] = b;
    __syncthreads();
    for (int s = 128; s > 0; s >>= 1) {
        if (threadIdx.x < s) { s1[threadIdx.x] += s1[threadIdx.x + s]; s2[threadIdx.x] += s2[threadIdx.x + s]; }
        __syncthreads();
    }
    if (threadIdx.x == 0) {
        float m = s1[0] / rows;
        mu[c] = m;
        var[c] = s2[0] / rows - m * m;
    }
}

// ---------------- pooling init / fused BN1-apply + segment sum/max/count ----------------
__global__ void k_pool_init(float* Hsum, float* Hmax, float* cnt) {
    int i = blockIdx.x * blockDim.x + threadIdx.x;
    if (i < GG * DOUT) { Hsum[i] = 0.f; Hmax[i] = NEG_BIG; }
    if (i < GG) cnt[i] = 0.f;
}
__device__ inline void atomicMaxF(float* p, float v) {
    unsigned* up = (unsigned*)p;
    unsigned old = *up;
    for (;;) {
        float f = __uint_as_float(old);
        if (f >= v) break;
        unsigned assumed = old;
        old = atomicCAS(up, assumed, __float_as_uint(v));
        if (old == assumed) break;
    }
}
__global__ void k_pool(const float* __restrict__ H, const long long* __restrict__ batch,
                       const float* __restrict__ mu, const float* __restrict__ var,
                       const float* __restrict__ gamma, const float* __restrict__ beta,
                       float* __restrict__ Hsum, float* __restrict__ Hmax, float* __restrict__ cnt) {
    int idx = blockIdx.x * blockDim.x + threadIdx.x;
    if (idx >= NN * DOUT) return;
    int n = idx / DOUT, c = idx % DOUT;
    int g = (int)batch[n];
    float v = gamma[c] * (H[idx] - mu[c]) * rsqrtf(var[c] + BNEPS) + beta[c];
    atomicAdd(&Hsum[g * DOUT + c], v);
    atomicMaxF(&Hmax[g * DOUT + c], v);
    if (c == 0) atomicAdd(&cnt[g], 1.0f);
}
__global__ void k_build_hg(const float* __restrict__ Hsum, const float* __restrict__ Hmax,
                           const float* __restrict__ cnt, float* __restrict__ Hg) {
    int idx = blockIdx.x * blockDim.x + threadIdx.x;
    if (idx >= GG * 3 * DOUT) return;
    int g = idx / (3 * DOUT), c = idx % (3 * DOUT);
    float v;
    if      (c < DOUT)     v = Hsum[g * DOUT + c] / fmaxf(cnt[g], 1.0f);   // avg
    else if (c < 2 * DOUT) v = Hsum[g * DOUT + (c - DOUT)];                 // sum
    else                   v = Hmax[g * DOUT + (c - 2 * DOUT)];             // max
    Hg[idx] = v;
}
// BN2 over GG rows, 576 channels, normalize in place (one thread per channel)
__global__ void k_bn2(float* __restrict__ Hg, const float* __restrict__ gamma,
                      const float* __restrict__ beta) {
    int c = blockIdx.x * blockDim.x + threadIdx.x;
    if (c >= 3 * DOUT) return;
    float a = 0.f, b = 0.f;
    for (int g = 0; g < GG; ++g) { float v = Hg[g * 3 * DOUT + c]; a += v; b += v * v; }
    float m = a / GG;
    float rv = rsqrtf(b / GG - m * m + BNEPS);
    for (int g = 0; g < GG; ++g) {
        float v = Hg[g * 3 * DOUT + c];
        Hg[g * 3 * DOUT + c] = gamma[c] * (v - m) * rv + beta[c];
    }
}

// ---------------- funnel head ----------------
__global__ void k_fc_relu(const float* __restrict__ in, const float* __restrict__ w,
                          const float* __restrict__ bias, float* __restrict__ out,
                          int IN, int OUT) {
    int idx = blockIdx.x * blockDim.x + threadIdx.x;
    if (idx >= GG * OUT) return;
    int g = idx / OUT, o = idx % OUT;
    float a = bias[o];
    const float* ip = in + (size_t)g * IN;
    const float* wp = w + (size_t)o * IN;
    for (int k = 0; k < IN; ++k) a = fmaf(ip[k], wp[k], a);
    out[idx] = fmaxf(a, 0.f);
}
__global__ void k_head(const float* __restrict__ in, const float* __restrict__ w3,
                       const float* __restrict__ b3, float* __restrict__ out) {
    int g = blockIdx.x * blockDim.x + threadIdx.x;
    if (g >= GG) return;
    float logits[NCL];
    float mx = NEG_BIG;
    for (int o = 0; o < NCL; ++o) {
        float a = b3[o];
        for (int k = 0; k < DOUT; ++k) a = fmaf(in[g * DOUT + k], w3[o * DOUT + k], a);
        logits[o] = a;
        mx = fmaxf(mx, a);
    }
    float s = 0.f;
    for (int o = 0; o < NCL; ++o) s += expf(logits[o] - mx);
    float lse = mx + logf(s);
    for (int o = 0; o < NCL; ++o) out[g * NCL + o] = logits[o] - lse;
}

// ---------------- launch ----------------
extern "C" void kernel_launch(void* const* d_in, const int* in_sizes, int n_in,
                              void* d_out, int out_size, void* d_ws, size_t ws_size,
                              hipStream_t stream) {
    const float*     x     = (const float*)d_in[0];
    const long long* ei    = (const long long*)d_in[1];
    const long long* batch = (const long long*)d_in[2];
    const float* W    = (const float*)d_in[3];
    const float* bvec = (const float*)d_in[4];
    const float* u    = (const float*)d_in[5];
    const float* g1   = (const float*)d_in[6];
    const float* bt1  = (const float*)d_in[7];
    const float* g2   = (const float*)d_in[8];
    const float* bt2  = (const float*)d_in[9];
    const float* w1   = (const float*)d_in[10];
    const float* b1   = (const float*)d_in[11];
    const float* w2   = (const float*)d_in[12];
    const float* b2   = (const float*)d_in[13];
    const float* w3   = (const float*)d_in[14];
    const float* b3   = (const float*)d_in[15];
    float* out = (float*)d_out;

    char* ws = (char*)d_ws;
    size_t cur = 0;
    auto alloc = [&](size_t bytes) -> char* {
        char* p = ws + cur;
        cur = (cur + bytes + 255) & ~(size_t)255;
        return p;
    };
    float*  deg   = (float*)alloc((size_t)NN * 4);
    float*  H1    = (float*)alloc((size_t)NN * INC * 4);
    float*  H2    = (float*)alloc((size_t)NN * INC * 4);
    __bf16* Abf   = (__bf16*)alloc((size_t)NN * DIN * 2);
    __bf16* Wbf   = (__bf16*)alloc((size_t)DOUT * DIN * 2);
    float*  isig  = (float*)alloc(256);
    float*  H     = (float*)alloc((size_t)NN * DOUT * 4);
    float*  mu    = (float*)alloc(DOUT * 4);
    float*  var   = (float*)alloc(DOUT * 4);
    float*  Hsum  = (float*)alloc((size_t)GG * DOUT * 4);
    float*  Hmax  = (float*)alloc((size_t)GG * DOUT * 4);
    float*  cnt   = (float*)alloc(GG * 4);
    float*  Hg    = (float*)alloc((size_t)GG * 3 * DOUT * 4);
    float*  xo1   = (float*)alloc((size_t)GG * 2 * DOUT * 4);
    float*  xo2   = (float*)alloc((size_t)GG * DOUT * 4);

    const int T = 256;
    // degree -> dinv
    k_zero  <<<(NN + T - 1) / T, T, 0, stream>>>(deg, NN);
    k_degree<<<(EE + T - 1) / T, T, 0, stream>>>(ei, deg);
    k_dinv  <<<(NN + T - 1) / T, T, 0, stream>>>(deg);
    // H1 = L x ; H2 = L H1   (sparse: identity copy + edge scatter)
    k_copy<<<(NN * INC + T - 1) / T, T, 0, stream>>>(x, H1, NN * INC);
    k_spmm<<<(EE + T - 1) / T, T, 0, stream>>>(ei, deg, x, H1);
    k_copy<<<(NN * INC + T - 1) / T, T, 0, stream>>>(H1, H2, NN * INC);
    k_spmm<<<(EE + T - 1) / T, T, 0, stream>>>(ei, deg, H1, H2);
    // spectral norm, masked weight, bf16 operands
    k_specnorm<<<1, DIN, 0, stream>>>(W, u, isig);
    k_build_wbf<<<(DOUT * DIN + T - 1) / T, T, 0, stream>>>(W, isig, Wbf);
    k_cat_bf16<<<(NN * DIN + T - 1) / T, T, 0, stream>>>(x, H1, H2, Abf);
    // masked linear via v_wmma_f32_16x16x32_bf16 with async-LDS B staging
    int blocks = (NN / 16 / WPB) * TILES_N;     // 128 * 12
    k_gemm_wmma<<<blocks, 32 * WPB, 0, stream>>>(Abf, Wbf, bvec, H);
    // BN1 + fused triple pooling
    k_bn_stats<<<DOUT, 256, 0, stream>>>(H, mu, var, NN, DOUT);
    k_pool_init<<<(GG * DOUT + T - 1) / T, T, 0, stream>>>(Hsum, Hmax, cnt);
    k_pool<<<(NN * DOUT + T - 1) / T, T, 0, stream>>>(H, batch, mu, var, g1, bt1, Hsum, Hmax, cnt);
    k_build_hg<<<(GG * 3 * DOUT + T - 1) / T, T, 0, stream>>>(Hsum, Hmax, cnt, Hg);
    k_bn2<<<(3 * DOUT + T - 1) / T, T, 0, stream>>>(Hg, g2, bt2);
    // funnel head
    k_fc_relu<<<(GG * 2 * DOUT + T - 1) / T, T, 0, stream>>>(Hg, w1, b1, xo1, 3 * DOUT, 2 * DOUT);
    k_fc_relu<<<(GG * DOUT + T - 1) / T, T, 0, stream>>>(xo1, w2, b2, xo2, 2 * DOUT, DOUT);
    k_head<<<1, GG, 0, stream>>>(xo2, w3, b3, out);
}
